// AttentionSequence_52003464020009
// MI455X (gfx1250) — matrix operand order repaired and checked by reading, not compiled, not run on hardware
//
#include <hip/hip_runtime.h>
#include <hip/hip_bf16.h>
#include <math.h>

// ---------------------------------------------------------------------------
// DIN attention (query/keys + DICE MLP + softmax pooling) for MI455X gfx1250.
// Phase 1: GEMM1 (bf16 WMMA, f32 acc) -> Z1 + batch stats (atomics)
// Phase 2: DICE1 + GEMM2 (bf16 WMMA)  -> Z2 + batch stats
// Phase 3: DICE2 + projection + masked stable softmax + weighted key sum
// v2: vectorized staging (b128 global loads, packed bf16 b128 LDS stores),
//     2 row-tiles per wave (B-fragment reuse), DICE affine params in LDS.
// ---------------------------------------------------------------------------

#define B_   2048
#define S_   200
#define E_   64
#define N_   (B_ * S_)      // 409600 rows
#define DIN_ 256            // 4*E
#define H1_  80
#define H2_  40
#define H2P_ 48             // H2 padded to 3 tiles of 16
#define K2P_ 96             // H1 padded to 3 K-chunks of 32
#define RPB_ 256            // rows per block (8 waves x 32 rows)

static __device__ constexpr float kEps = 1e-5f;

typedef __bf16 bf16_t;
typedef __attribute__((ext_vector_type(16))) __bf16 bf16x16;
typedef __attribute__((ext_vector_type(8)))  __bf16 bf16x8;
typedef __attribute__((ext_vector_type(8)))  float  floatx8;
typedef __attribute__((ext_vector_type(4)))  float  floatx4;

union FragAB { bf16x16 v; bf16x8 h[2]; };

__device__ __forceinline__ bf16x8 pack8(floatx4 v0, floatx4 v1) {
    bf16x8 p;
    p[0] = (bf16_t)v0[0]; p[1] = (bf16_t)v0[1];
    p[2] = (bf16_t)v0[2]; p[3] = (bf16_t)v0[3];
    p[4] = (bf16_t)v1[0]; p[5] = (bf16_t)v1[1];
    p[6] = (bf16_t)v1[2]; p[7] = (bf16_t)v1[3];
    return p;
}

// ---------------------------------------------------------------------------
// Kernel 0: zero the stats accumulators (2*H1 + 2*H2 floats, contiguous).
// ---------------------------------------------------------------------------
__global__ void init_stats_kernel(float* __restrict__ stats) {
    int t = threadIdx.x;
    if (t < 2 * H1_ + 2 * H2_) stats[t] = 0.0f;
}

// ---------------------------------------------------------------------------
// Kernel 1: Z1 = att_input @ W1 + b1, att_input built on the fly.
// 256 rows/block, 8 waves, each wave = 32 rows (2 A-tiles) x 80 cols
// (5 B-tiles reused across both row tiles), K in 32-wide bf16 chunks.
// ---------------------------------------------------------------------------
__global__ __launch_bounds__(256)
void gemm1_stats_kernel(const float* __restrict__ query,
                        const float* __restrict__ keys,
                        const float* __restrict__ W1,
                        const float* __restrict__ b1,
                        float* __restrict__ z1,
                        float* __restrict__ stats1)   // [2*H1]
{
    __shared__ __align__(16) bf16_t Wt[H1_ * DIN_];   // W1^T [n][k], 40KB
    __shared__ __align__(16) bf16_t Abuf[RPB_ * 32];  // A chunk, 16KB
    __shared__ float sbuf[2 * H1_];

    const int tid  = threadIdx.x;
    const int wave = tid >> 5;
    const int lane = tid & 31;
    const int m    = lane & 15;
    const bool hi  = lane >= 16;
    const int r0   = blockIdx.x * RPB_;

    for (int idx = tid; idx < H1_ * DIN_; idx += 256) {
        int n = idx >> 8;            // / 256
        int k = idx & 255;
        Wt[idx] = (bf16_t)W1[k * H1_ + n];
    }
    for (int idx = tid; idx < 2 * H1_; idx += 256) sbuf[idx] = 0.0f;
    __syncthreads();

    floatx8 acc[2][5] = {};

    for (int kk = 0; kk < DIN_; kk += 32) {
        const int seg = kk >> 6;     // concat segment: q | k | q-k | q*k
        const int eb  = kk & 63;

        // Stage att_input chunk [256 x 32] bf16: 8 e-values per unit,
        // two b128 global loads + one b128 LDS store.
        #pragma unroll
        for (int t = 0; t < 4; ++t) {
            int u  = t * 256 + tid;
            int i  = u >> 2;          // local row
            int g  = u & 3;           // e-group of 8
            int n  = r0 + i;
            int e0 = eb + g * 8;
            floatx4 v0, v1;
            if (seg == 0) {
                const floatx4* q = (const floatx4*)&query[(n / S_) * E_ + e0];
                v0 = q[0]; v1 = q[1];
            } else if (seg == 1) {
                const floatx4* kp = (const floatx4*)&keys[(size_t)n * E_ + e0];
                v0 = kp[0]; v1 = kp[1];
            } else {
                const floatx4* q  = (const floatx4*)&query[(n / S_) * E_ + e0];
                const floatx4* kp = (const floatx4*)&keys[(size_t)n * E_ + e0];
                if (seg == 2) { v0 = q[0] - kp[0]; v1 = q[1] - kp[1]; }
                else          { v0 = q[0] * kp[0]; v1 = q[1] * kp[1]; }
            }
            *(bf16x8*)&Abuf[i * 32 + g * 8] = pack8(v0, v1);
        }
        __syncthreads();

        // A fragments: 16x32 bf16 per ISA layout (lanes<16: K{0-7,16-23}).
        const int ka = hi ? 8 : 0;
        const int kb = kk + (hi ? 8 : 0);
        FragAB a[2];
        #pragma unroll
        for (int rt = 0; rt < 2; ++rt) {
            const int arow = (wave * 32 + rt * 16 + m) * 32;
            a[rt].h[0] = *(const bf16x8*)&Abuf[arow + ka];
            a[rt].h[1] = *(const bf16x8*)&Abuf[arow + ka + 16];
        }
        #pragma unroll
        for (int c = 0; c < 5; ++c) {
            FragAB bm;                       // B: same K layout as A
            const int brow = (c * 16 + m) * DIN_;
            bm.h[0] = *(const bf16x8*)&Wt[brow + kb];
            bm.h[1] = *(const bf16x8*)&Wt[brow + kb + 16];
            acc[0][c] = __builtin_amdgcn_wmma_f32_16x16x32_bf16(
                false, a[0].v, false, bm.v, (short)0, acc[0][c], false, false);
            acc[1][c] = __builtin_amdgcn_wmma_f32_16x16x32_bf16(
                false, a[1].v, false, bm.v, (short)0, acc[1][c], false, false);
        }
        __syncthreads();
    }

    // Epilogue: bias, store Z1, per-feature sum / sumsq reduction.
    #pragma unroll
    for (int rt = 0; rt < 2; ++rt) {
        #pragma unroll
        for (int c = 0; c < 5; ++c) {
            const int col = c * 16 + m;
            const float bias = b1[col];
            float s1 = 0.0f, s2 = 0.0f;
            #pragma unroll
            for (int r = 0; r < 8; ++r) {
                const int rowl = wave * 32 + rt * 16 + (hi ? 8 + r : r);
                const float z = acc[rt][c][r] + bias;
                z1[(size_t)(r0 + rowl) * H1_ + col] = z;
                s1 += z;
                s2 += z * z;
            }
            atomicAdd(&sbuf[col], s1);
            atomicAdd(&sbuf[H1_ + col], s2);
        }
    }
    __syncthreads();
    for (int idx = tid; idx < 2 * H1_; idx += 256)
        atomicAdd(&stats1[idx], sbuf[idx]);
}

// ---------------------------------------------------------------------------
// Kernel 2: h1 = DICE(Z1) (global batch stats), Z2 = h1 @ W2 + b2.
// DICE folded to xn = x*dA[k] + dC[k]; params precomputed in LDS.
// K padded 80->96, cols padded 40->48 (3 tiles).
// ---------------------------------------------------------------------------
__global__ __launch_bounds__(256)
void gemm2_stats_kernel(const float* __restrict__ z1,
                        const float* __restrict__ stats1,
                        const float* __restrict__ gamma1,
                        const float* __restrict__ beta1,
                        const float* __restrict__ alpha1,
                        const float* __restrict__ W2,
                        const float* __restrict__ b2,
                        float* __restrict__ z2,
                        float* __restrict__ stats2)   // [2*H2]
{
    __shared__ __align__(16) bf16_t Wt[H2P_ * K2P_];  // [48][96] zero-padded
    __shared__ __align__(16) bf16_t Abuf[RPB_ * 32];  // 16KB
    __shared__ float sbuf[2 * H2_];
    __shared__ float dA[H1_], dC[H1_], dAl[H1_];

    const int tid  = threadIdx.x;
    const int wave = tid >> 5;
    const int lane = tid & 31;
    const int m    = lane & 15;
    const bool hi  = lane >= 16;
    const int r0   = blockIdx.x * RPB_;
    const float invN = 1.0f / (float)N_;

    if (tid < H1_) {
        float mu = stats1[tid] * invN;
        float vr = stats1[H1_ + tid] * invN - mu * mu;
        float aa = rsqrtf(vr + kEps) * gamma1[tid];
        dA[tid]  = aa;
        dC[tid]  = beta1[tid] - mu * aa;
        dAl[tid] = alpha1[tid];
    }
    for (int idx = tid; idx < H2P_ * K2P_; idx += 256) {
        int n = idx / K2P_;
        int k = idx % K2P_;
        float w = (n < H2_ && k < H1_) ? W2[k * H2_ + n] : 0.0f;
        Wt[idx] = (bf16_t)w;
    }
    for (int idx = tid; idx < 2 * H2_; idx += 256) sbuf[idx] = 0.0f;
    __syncthreads();

    floatx8 acc[2][3] = {};

    for (int kk = 0; kk < K2P_; kk += 32) {
        #pragma unroll
        for (int t = 0; t < 4; ++t) {
            int u  = t * 256 + tid;
            int i  = u >> 2;
            int g  = u & 3;
            int k0 = kk + g * 8;
            bf16x8 pk = {};
            if (k0 < H1_) {                       // whole 8-group valid
                int n = r0 + i;
                const floatx4* zp = (const floatx4*)&z1[(size_t)n * H1_ + k0];
                floatx4 x0 = zp[0], x1 = zp[1];
                floatx4 h0, h1;
                #pragma unroll
                for (int jj = 0; jj < 4; ++jj) {
                    int k = k0 + jj;
                    float x  = x0[jj];
                    float xn = fmaf(x, dA[k], dC[k]);
                    float p  = 1.0f / (1.0f + __expf(-xn));
                    h0[jj] = x * (p + dAl[k] * (1.0f - p));
                }
                #pragma unroll
                for (int jj = 0; jj < 4; ++jj) {
                    int k = k0 + 4 + jj;
                    float x  = x1[jj];
                    float xn = fmaf(x, dA[k], dC[k]);
                    float p  = 1.0f / (1.0f + __expf(-xn));
                    h1[jj] = x * (p + dAl[k] * (1.0f - p));
                }
                pk = pack8(h0, h1);
            }
            *(bf16x8*)&Abuf[i * 32 + g * 8] = pk;
        }
        __syncthreads();

        const int ka = hi ? 8 : 0;
        const int kb = kk + (hi ? 8 : 0);
        FragAB a[2];
        #pragma unroll
        for (int rt = 0; rt < 2; ++rt) {
            const int arow = (wave * 32 + rt * 16 + m) * 32;
            a[rt].h[0] = *(const bf16x8*)&Abuf[arow + ka];
            a[rt].h[1] = *(const bf16x8*)&Abuf[arow + ka + 16];
        }
        #pragma unroll
        for (int c = 0; c < 3; ++c) {
            FragAB bm;
            const int brow = (c * 16 + m) * K2P_;
            bm.h[0] = *(const bf16x8*)&Wt[brow + kb];
            bm.h[1] = *(const bf16x8*)&Wt[brow + kb + 16];
            acc[0][c] = __builtin_amdgcn_wmma_f32_16x16x32_bf16(
                false, a[0].v, false, bm.v, (short)0, acc[0][c], false, false);
            acc[1][c] = __builtin_amdgcn_wmma_f32_16x16x32_bf16(
                false, a[1].v, false, bm.v, (short)0, acc[1][c], false, false);
        }
        __syncthreads();
    }

    #pragma unroll
    for (int rt = 0; rt < 2; ++rt) {
        #pragma unroll
        for (int c = 0; c < 3; ++c) {
            const int col = c * 16 + m;
            if (col < H2_) {
                const float bias = b2[col];
                float s1 = 0.0f, s2 = 0.0f;
                #pragma unroll
                for (int r = 0; r < 8; ++r) {
                    const int rowl = wave * 32 + rt * 16 + (hi ? 8 + r : r);
                    const float z = acc[rt][c][r] + bias;
                    z2[(size_t)(r0 + rowl) * H2_ + col] = z;
                    s1 += z;
                    s2 += z * z;
                }
                atomicAdd(&sbuf[col], s1);
                atomicAdd(&sbuf[H2_ + col], s2);
            }
        }
    }
    __syncthreads();
    for (int idx = tid; idx < 2 * H2_; idx += 256)
        atomicAdd(&stats2[idx], sbuf[idx]);
}

// ---------------------------------------------------------------------------
// Kernel 3: per-batch scores (DICE2 + projection), mask, stable softmax over
// S=200, then user_interest[b,e] = sum_s w[s] * keys[b,s,e]. One block per b.
// ---------------------------------------------------------------------------
__global__ __launch_bounds__(256)
void attend_kernel(const float* __restrict__ keys,
                   const float* __restrict__ z2,
                   const float* __restrict__ stats2,
                   const float* __restrict__ gamma2,
                   const float* __restrict__ beta2,
                   const float* __restrict__ alpha2,
                   const float* __restrict__ Wp,
                   const float* __restrict__ bp,
                   float* __restrict__ out)
{
    __shared__ float sc[256];
    __shared__ float red[256];
    __shared__ float psum[4][64];
    __shared__ float dA[H2_], dC[H2_], dW[H2_];   // DICE2 affine params + plain copy of Wp

    const int tid = threadIdx.x;
    const int b   = blockIdx.x;
    const float invN = 1.0f / (float)N_;

    if (tid < H2_) {
        float mu = stats2[tid] * invN;
        float vr = stats2[H2_ + tid] * invN - mu * mu;
        float aa = rsqrtf(vr + kEps) * gamma2[tid];
        dA[tid] = aa;
        dC[tid] = beta2[tid] - mu * aa;
        dW[tid] = Wp[tid];
    }
    __syncthreads();

    float score = -INFINITY;
    if (tid < S_) {
        const int n = b * S_ + tid;
        float acc = bp[0];
        const floatx4* zp = (const floatx4*)&z2[(size_t)n * H2_];
        #pragma unroll
        for (int g = 0; g < H2_ / 4; ++g) {
            floatx4 xv = zp[g];
            #pragma unroll
            for (int jj = 0; jj < 4; ++jj) {
                int j = g * 4 + jj;
                float x  = xv[jj];
                float xn = fmaf(x, dA[j], dC[j]);
                float p  = 1.0f / (1.0f + __expf(-xn));
                acc += x * (p + alpha2[j] * (1.0f - p)) * dW[j];
            }
        }
        float ks = 0.0f;
        const floatx4* kp = (const floatx4*)&keys[(size_t)n * E_];
        #pragma unroll
        for (int g = 0; g < E_ / 4; ++g) {
            floatx4 kv = kp[g];
            ks += kv[0] + kv[1] + kv[2] + kv[3];
        }
        score = (ks != 0.0f) ? acc : -1e9f;
    }
    sc[tid]  = score;
    red[tid] = score;
    __syncthreads();

    for (int off = 128; off > 0; off >>= 1) {      // max reduce
        if (tid < off) red[tid] = fmaxf(red[tid], red[tid + off]);
        __syncthreads();
    }
    const float mx = red[0];
    __syncthreads();

    float ex = (tid < S_) ? __expf(sc[tid] - mx) : 0.0f;
    red[tid] = ex;
    __syncthreads();
    for (int off = 128; off > 0; off >>= 1) {      // sum reduce
        if (tid < off) red[tid] += red[tid + off];
        __syncthreads();
    }
    const float inv_sum = 1.0f / red[0];
    __syncthreads();
    sc[tid] = ex * inv_sum;                        // softmax weights
    __syncthreads();

    // Weighted sum of keys: 4 S-chunks x 64 features.
    const int e = tid & 63;
    const int c = tid >> 6;
    float part = 0.0f;
    const int s0 = c * 50;
    for (int s = s0; s < s0 + 50; ++s)
        part += sc[s] * keys[((size_t)(b * S_ + s)) * E_ + e];
    psum[c][e] = part;
    __syncthreads();
    if (tid < 64)
        out[b * E_ + tid] = psum[0][tid] + psum[1][tid] + psum[2][tid] + psum[3][tid];
}

// ---------------------------------------------------------------------------
extern "C" void kernel_launch(void* const* d_in, const int* in_sizes, int n_in,
                              void* d_out, int out_size, void* d_ws, size_t ws_size,
                              hipStream_t stream)
{
    const float* query  = (const float*)d_in[0];
    const float* keys   = (const float*)d_in[1];
    const float* W1     = (const float*)d_in[2];
    const float* b1     = (const float*)d_in[3];
    const float* gamma1 = (const float*)d_in[4];
    const float* beta1  = (const float*)d_in[5];
    const float* alpha1 = (const float*)d_in[6];
    const float* W2     = (const float*)d_in[7];
    const float* b2     = (const float*)d_in[8];
    const float* gamma2 = (const float*)d_in[9];
    const float* beta2  = (const float*)d_in[10];
    const float* alpha2 = (const float*)d_in[11];
    const float* Wp     = (const float*)d_in[12];
    const float* bp     = (const float*)d_in[13];
    float* out = (float*)d_out;

    // Workspace layout: Z1 [N,80] f32 | Z2 [N,40] f32 | stats1[160] | stats2[80]
    float* z1     = (float*)d_ws;
    float* z2     = z1 + (size_t)N_ * H1_;
    float* stats1 = z2 + (size_t)N_ * H2_;
    float* stats2 = stats1 + 2 * H1_;

    init_stats_kernel<<<dim3(1), dim3(256), 0, stream>>>(stats1);
    gemm1_stats_kernel<<<dim3(N_ / RPB_), dim3(256), 0, stream>>>(
        query, keys, W1, b1, z1, stats1);
    gemm2_stats_kernel<<<dim3(N_ / RPB_), dim3(256), 0, stream>>>(
        z1, stats1, gamma1, beta1, alpha1, W2, b2, z2, stats2);
    attend_kernel<<<dim3(B_), dim3(256), 0, stream>>>(
        keys, z2, stats2, gamma2, beta2, alpha2, Wp, bp, out);
}
